// WaterNetCQ1116_38268158608160
// MI455X (gfx1250) — matrix operand order repaired and checked by reading, not compiled, not run on hardware
//
#include <hip/hip_runtime.h>
#include <hip/hip_bf16.h>
#include <math.h>

#define NHID 16
#define NGRP 32
#define NRT  15
#define NTT  1000
#define NSS  1000
#define NLL  (NTT - NRT + 1)   /* 986 */
#define NROW (NTT * NSS)
#define TPB_TILES 25           /* row-tiles processed per block in the WMMA kernel */

typedef __attribute__((ext_vector_type(16))) _Float16 v16h;
typedef __attribute__((ext_vector_type(8)))  float    v8f;

__device__ __forceinline__ float sigmoidf_(float v) { return 1.0f / (1.0f + expf(-v)); }

// Branch-free tanh on the HW transcendental path:
// tanh(x) = copysign(1 - 2*rcp(exp(2|x|)+1), x)
// exp -> v_exp_f32 (saturates to +inf for large |x| => result +/-1), rcp -> v_rcp_f32.
__device__ __forceinline__ float fast_tanhf(float v)
{
  const float a = __builtin_fabsf(v);
  const float e = __expf(2.0f * a);
  const float r = 1.0f - 2.0f * __builtin_amdgcn_rcpf(e + 1.0f);
  return __builtin_copysignf(r, v);
}

// ---------------------------------------------------------------------------
// Kernel 1: per-site small MLPs (fc, fcR) + derived parameters
// ---------------------------------------------------------------------------
__global__ void __launch_bounds__(256) site_params_kernel(
    const float* __restrict__ xc,
    const float* __restrict__ fc_w1,  const float* __restrict__ fc_b1,
    const float* __restrict__ fc_w2,  const float* __restrict__ fc_b2,
    const float* __restrict__ fcR_w1, const float* __restrict__ fcR_b1,
    const float* __restrict__ fcR_w2, const float* __restrict__ fcR_b2,
    float* __restrict__ pk1,  float* __restrict__ pk2,  float* __restrict__ pk23,
    float* __restrict__ pk3,  float* __restrict__ pgl,  float* __restrict__ pqb,
    float* __restrict__ pge1, float* __restrict__ pge2, float* __restrict__ pga,
    float* __restrict__ prw)
{
  __shared__ float xcs[NGRP];
  __shared__ float hbuf[256];
  __shared__ float wbuf[160];
  __shared__ float wRbuf[240];

  const int s   = blockIdx.x;
  const int tid = threadIdx.x;

  if (tid < NGRP) xcs[tid] = xc[s * NGRP + tid];
  __syncthreads();

  // fc hidden: 256 units, tanh
  {
    float acc = fc_b1[tid];
    for (int k = 0; k < NGRP; ++k) acc += xcs[k] * fc_w1[k * 256 + tid];
    hbuf[tid] = tanhf(acc);
  }
  __syncthreads();
  if (tid < 160) {
    float acc = fc_b2[tid];
    for (int k = 0; k < 256; ++k) acc += hbuf[k] * fc_w2[k * 160 + tid];
    wbuf[tid] = acc;
  }
  __syncthreads();

  // fcR hidden (reuse hbuf)
  {
    float acc = fcR_b1[tid];
    for (int k = 0; k < NGRP; ++k) acc += xcs[k] * fcR_w1[k * 256 + tid];
    hbuf[tid] = tanhf(acc);
  }
  __syncthreads();
  if (tid < 240) {
    float acc = fcR_b2[tid];
    for (int k = 0; k < 256; ++k) acc += hbuf[k] * fcR_w2[k * 240 + tid];
    wRbuf[tid] = acc;
  }
  __syncthreads();

  if (tid < NHID) {
    const int h    = tid;
    const int base = s * NHID + h;
    pk1[base]  = sigmoidf_(wbuf[1 * NHID + h]);
    pk2[base]  = sigmoidf_(wbuf[2 * NHID + h]);
    pk23[base] = sigmoidf_(wbuf[3 * NHID + h]);
    pk3[base]  = sigmoidf_(wbuf[4 * NHID + h]) * 0.1f;
    pgl[base]  = expf(wbuf[5 * NHID + h]) * 2.0f;
    pqb[base]  = fmaxf(wbuf[7 * NHID + h], 0.0f);
    pge1[base] = fmaxf(wbuf[8 * NHID + h], 0.0f);
    pge2[base] = fmaxf(wbuf[9 * NHID + h], 0.0f);

    // ga = softmax over the 16 hidden buckets (redundant per-thread reduce)
    float mx = wbuf[6 * NHID];
    for (int j = 1; j < NHID; ++j) mx = fmaxf(mx, wbuf[6 * NHID + j]);
    float sum = 0.0f;
    for (int j = 0; j < NHID; ++j) sum += expf(wbuf[6 * NHID + j] - mx);
    pga[base] = expf(wbuf[6 * NHID + h] - mx) / sum;

    // rw = softmax(relu(wR)) over the 15 routing taps of bucket h
    float rv[NRT];
    float mxr = 0.0f;
    for (int j = 0; j < NRT; ++j) {
      rv[j] = fmaxf(wRbuf[h * NRT + j], 0.0f);
      mxr   = fmaxf(mxr, rv[j]);
    }
    float sr = 0.0f;
    for (int j = 0; j < NRT; ++j) { rv[j] = expf(rv[j] - mxr); sr += rv[j]; }
    for (int j = 0; j < NRT; ++j) prw[(h * NRT + j) * NSS + s] = rv[j] / sr;
  }
}

// ---------------------------------------------------------------------------
// Kernel 1b: hoist the rain/snow partition (acos) out of the sequential scan.
// Fully parallel over 10^6 (t,s) points; writes Ps, Pl.
// ---------------------------------------------------------------------------
__global__ void __launch_bounds__(256) pspl_kernel(
    const float* __restrict__ x, float* __restrict__ PsA, float* __restrict__ PlA)
{
  const int r = blockIdx.x * 256 + threadIdx.x;
  if (r >= NROW) return;
  const float P  = x[r * 6 + 0];
  const float T1 = x[r * 6 + 2];
  const float T2 = x[r * 6 + 3];
  const float den   = (fabsf(T2 - T1) > 1e-6f) ? (T2 - T1) : 1.0f;
  const float arg   = fminf(fmaxf((T1 + T2) / den, -1.0f + 1e-6f), 1.0f - 1e-6f);
  const float basev = 1.0f - acosf(arg) / 3.1415f;
  const float vp    = (T2 <= 0.0f) ? 0.0f : ((T1 >= 0.0f) ? 1.0f : basev);
  PsA[r] = P * (1.0f - vp);
  PlA[r] = P * vp;
}

// ---------------------------------------------------------------------------
// WMMA fragment loaders (CDNA5 §7.12.2 layouts, f16, 16x16x32)
// ---------------------------------------------------------------------------
__device__ __forceinline__ v16h load_a_frag(const _Float16* __restrict__ base,
                                            int stride, int kOff, int lane)
{
  // A 16x32: lanes 0-15 = rows, K {0..7,16..23}; lanes 16-31 same rows, K {8..15,24..31}
  v16h a;
  const int m  = lane & 15;
  const int hi = lane >> 4;
  const _Float16* p = base + m * stride + kOff + hi * 8;
#pragma unroll
  for (int e = 0; e < 8; ++e) a[e] = p[e];
#pragma unroll
  for (int e = 0; e < 8; ++e) a[8 + e] = p[16 + e];
  return a;
}

__device__ __forceinline__ v16h load_b_frag(const _Float16* __restrict__ baseT,
                                            int stride, int n0, int kOff, int lane)
{
  // B 32x16 from a transposed [n][k] tile: lanes 0-15 hold K {0..7,16..23}, lanes 16-31 K {8..15,24..31}
  v16h b;
  const int n  = lane & 15;
  const int kb = kOff + ((lane >> 4) << 3);
  const _Float16* p = baseT + (n0 + n) * stride + kb;
#pragma unroll
  for (int e = 0; e < 8; ++e) b[e] = p[e];
#pragma unroll
  for (int e = 0; e < 8; ++e) b[8 + e] = p[16 + e];
  return b;
}

// ---------------------------------------------------------------------------
// Kernel 2: batched MLPs (fcT1 / fcT2) via v_wmma_f32_16x16x32_f16.
// grid = (62500/TPB_TILES, 2), block = 256 (8 wave32).
// Weights staged in LDS ONCE per block, then TPB_TILES row-tiles streamed
// through, so staging cost is amortized 25x and blocks are WMMA-dense.
// ---------------------------------------------------------------------------
__global__ void __launch_bounds__(256) mlp_big_kernel(
    const float* __restrict__ x,  const float* __restrict__ xc,
    const float* __restrict__ w1a, const float* __restrict__ b1a,
    const float* __restrict__ w2a, const float* __restrict__ b2a,
    const float* __restrict__ w1b, const float* __restrict__ b1b,
    const float* __restrict__ w2b, const float* __restrict__ b2b,
    _Float16* __restrict__ vi_out, _Float16* __restrict__ vk_out,
    _Float16* __restrict__ vm_out)
{
  const int which    = blockIdx.y;       // 0 = fcT1, 1 = fcT2
  const int tileBase = blockIdx.x * TPB_TILES;
  const int tid      = threadIdx.x;
  const int lane     = tid & 31;
  const int wave     = tid >> 5;

  // padded strides (72/264 halves) keep 16B alignment and dodge bank conflicts
  __shared__ _Float16 Alds[16 * 72];    // 16 x 64 input tile
  __shared__ _Float16 W1T[256 * 72];    // layer-1 weights, [n][k0..63], zero-padded K
  __shared__ _Float16 Hlds[16 * 264];   // 16 x 256 hidden (f16)
  __shared__ _Float16 W2T[32 * 264];    // layer-2 weights, [n][k0..255]

  const float* w1 = which ? w1b : w1a;
  const float* b1 = which ? b1b : b1a;
  const float* w2 = which ? w2b : w2a;
  const float* b2 = which ? b2b : b2a;
  const int K1   = which ? 35 : 33;     // layer-1 input features
  const int N2   = which ? 16 : 32;     // layer-2 outputs we actually need
  const int N2st = which ? 17 : 32;     // row stride of the (256,Nout) weight

  // ---- stage weights once per block ----
  {
    const int n = tid;
    for (int k = 0; k < 64; ++k)
      W1T[n * 72 + k] = (k < K1) ? (_Float16)w1[k * 256 + n] : (_Float16)0.0f;
  }
  for (int idx = tid; idx < N2 * 256; idx += 256) {
    const int n = idx >> 8, k = idx & 255;
    W2T[n * 264 + k] = (_Float16)w2[k * N2st + n];
  }

  for (int it = 0; it < TPB_TILES; ++it) {
    const int rowTile = tileBase + it;

    // prefetch next tile's x rows (gfx1250 global_prefetch path)
    if (tid < 16 && it + 1 < TPB_TILES) {
      const int rn = (rowTile + 1) * 16 + tid;
      __builtin_prefetch(&x[rn * 6], 0, 3);
    }

    // stage the 16x64 A tile: [extras, xc] per row, zero padded
    for (int idx = tid; idx < 16 * 64; idx += 256) {
      const int m = idx >> 6, k = idx & 63;
      const int r = rowTile * 16 + m;
      const int s = r % NSS;
      float v = 0.0f;
      if (which == 0) {
        if (k == 0)        v = x[r * 6 + 5];            // LAI
        else if (k <= 32)  v = xc[s * NGRP + (k - 1)];
      } else {
        if (k == 0)        v = x[r * 6 + 4];            // R
        else if (k == 1)   v = x[r * 6 + 2];            // T1
        else if (k == 2)   v = x[r * 6 + 3];            // T2
        else if (k <= 34)  v = xc[s * NGRP + (k - 3)];
      }
      Alds[m * 72 + k] = (_Float16)v;
    }
    __syncthreads();   // A (and, on iter 0, weights) visible

    // ---- layer 1: (16x64) @ (64x256) -> tanh -> Hlds, each wave owns 32 cols ----
    const v16h a0 = load_a_frag(Alds, 72, 0, lane);
    const v16h a1 = load_a_frag(Alds, 72, 32, lane);
#pragma unroll
    for (int sub = 0; sub < 2; ++sub) {
      const int n0 = wave * 32 + sub * 16;
      v8f acc;
      const float bias = b1[n0 + (lane & 15)];
#pragma unroll
      for (int i = 0; i < 8; ++i) acc[i] = bias;
      const v16h bf0 = load_b_frag(W1T, 72, n0, 0, lane);
      acc = __builtin_amdgcn_wmma_f32_16x16x32_f16(false, a0, false, bf0,
                                                   (short)0, acc, false, false);
      const v16h bf1 = load_b_frag(W1T, 72, n0, 32, lane);
      acc = __builtin_amdgcn_wmma_f32_16x16x32_f16(false, a1, false, bf1,
                                                   (short)0, acc, false, false);
      // C layout: elem i -> M = i + 8*(lane>>4), N = n0 + (lane&15)
      const int nn    = n0 + (lane & 15);
      const int mbase = (lane >> 4) * 8;
#pragma unroll
      for (int i = 0; i < 8; ++i)
        Hlds[(mbase + i) * 264 + nn] = (_Float16)fast_tanhf(acc[i]);
    }
    __syncthreads();   // H ready

    // ---- layer 2: (16x256) @ (256xN2), one wave per 16-col tile ----
    if (wave < (N2 >> 4)) {
      const int n0 = wave * 16;
      v8f acc;
      const float bias = b2[n0 + (lane & 15)];
#pragma unroll
      for (int i = 0; i < 8; ++i) acc[i] = bias;
#pragma unroll
      for (int kc = 0; kc < 8; ++kc) {
        const v16h ah = load_a_frag(Hlds, 264, kc * 32, lane);
        const v16h bh = load_b_frag(W2T, 264, n0, kc * 32, lane);
        acc = __builtin_amdgcn_wmma_f32_16x16x32_f16(false, ah, false, bh,
                                                     (short)0, acc, false, false);
      }
      const int h     = lane & 15;
      const int mbase = (lane >> 4) * 8;
#pragma unroll
      for (int i = 0; i < 8; ++i) {
        const int r = rowTile * 16 + mbase + i;
        const float v = acc[i];
        if (which == 0) {
          const float hs = fminf(fmaxf(v * (1.0f / 6.0f) + 0.5f, 0.0f), 1.0f);
          if (wave == 0) vi_out[r * NHID + h] = (_Float16)hs;   // v1[:, :16]  -> vi
          else           vk_out[r * NHID + h] = (_Float16)hs;   // v1[:,16:32] -> vk
        } else {
          // vm = exp(2*v2) computed here (parallel) instead of in the scan loop
          vm_out[r * NHID + h] = (_Float16)__expf(2.0f * v);
        }
      }
    }
    __syncthreads();   // A/H fully consumed before next iteration overwrites
  }
}

// ---------------------------------------------------------------------------
// Kernel 3: sequential time scan; one thread per (site, bucket)
// writes Qg[t][h][s] = (Q1+Q2+Q3)*ga. Critical path: cheap VALU + one __powf.
// ---------------------------------------------------------------------------
__global__ void __launch_bounds__(256) scan_kernel(
    const float* __restrict__ x,
    const float* __restrict__ PsA, const float* __restrict__ PlA,
    const _Float16* __restrict__ vi_arr, const _Float16* __restrict__ vk_arr,
    const _Float16* __restrict__ vm_arr,
    const float* __restrict__ pk1,  const float* __restrict__ pk2,
    const float* __restrict__ pk23, const float* __restrict__ pk3,
    const float* __restrict__ pgl,  const float* __restrict__ pqb,
    const float* __restrict__ pge1, const float* __restrict__ pge2,
    const float* __restrict__ pga,  float* __restrict__ Qg)
{
  const int tid = blockIdx.x * blockDim.x + threadIdx.x;
  const int s = tid >> 4;
  const int h = tid & 15;
  if (s >= NSS) return;

  const int base = s * NHID + h;
  const float k1 = pk1[base],  k2 = pk2[base], k23 = pk23[base], k3 = pk3[base];
  const float gl = pgl[base],  qb = pqb[base];
  const float ge1 = pge1[base], ge2 = pge2[base], ga = pga[base];

  float S0 = 0.0f, Sv = 0.0f, S2 = 0.0f, S3 = 0.0f;
  for (int t = 0; t < NTT; ++t) {
    const int r = t * NSS + s;
    const float Ps = PsA[r];
    const float Pl = PlA[r];
    const float E  = x[r * 6 + 1];

    const float vi = (float)vi_arr[r * NHID + h];
    const float vk = (float)vk_arr[r * NHID + h];
    const float vm = (float)vm_arr[r * NHID + h];
    const float Pl1 = Pl * (1.0f - vi);
    const float Pl2 = Pl * vi;
    const float Ev1 = E * ge1;
    const float Ev2 = E * ge2;

    const float H0  = S0 + Ps;
    const float qSm = fminf(H0, vm);
    const float Hv  = fmaxf(Sv + Pl1 - Ev1, 0.0f);
    const float qv  = Sv * vk;
    const float H2  = fmaxf(S2 + qSm + qv - Ev2 + Pl2, 0.0f);
    const float hc  = fmaxf(H2 - gl, 0.0f);
    const float Q1  = (hc > 0.0f) ? __powf(hc, k1) : 0.0f;
    const float q2  = fminf(H2, gl) * k2;
    const float Q2  = q2 * (1.0f - k23);
    const float H3  = S3 + q2 * k23;
    const float Q3  = H3 * k3 + qb;

    S0 = H0 - qSm;
    Sv = Hv - qv;
    S2 = H2 - Q1 - q2;
    S3 = H3 - Q3;

    Qg[t * (NSS * NHID) + h * NSS + s] = (Q1 + Q2 + Q3) * ga;
  }
}

// ---------------------------------------------------------------------------
// Kernel 4: 15-tap depthwise valid conv + bucket reduction (Qg is L2-resident)
// ---------------------------------------------------------------------------
__global__ void __launch_bounds__(256) conv_kernel(
    const float* __restrict__ Qg, const float* __restrict__ rw,
    float* __restrict__ out)
{
  const int idx = blockIdx.x * blockDim.x + threadIdx.x;
  if (idx >= NLL * NSS) return;
  const int i = idx / NSS;
  const int s = idx % NSS;
  float acc = 0.0f;
  for (int h = 0; h < NHID; ++h) {
#pragma unroll
    for (int j = 0; j < NRT; ++j) {
      acc += Qg[(i + j) * (NSS * NHID) + h * NSS + s] * rw[(h * NRT + j) * NSS + s];
    }
  }
  out[i * NSS + s] = acc;
}

// ---------------------------------------------------------------------------
extern "C" void kernel_launch(void* const* d_in, const int* in_sizes, int n_in,
                              void* d_out, int out_size, void* d_ws, size_t ws_size,
                              hipStream_t stream)
{
  (void)in_sizes; (void)n_in; (void)out_size; (void)ws_size;

  const float* x  = (const float*)d_in[0];
  const float* xc = (const float*)d_in[1];
  const float* fc_w1  = (const float*)d_in[2];
  const float* fc_b1  = (const float*)d_in[3];
  const float* fc_w2  = (const float*)d_in[4];
  const float* fc_b2  = (const float*)d_in[5];
  const float* fcR_w1 = (const float*)d_in[6];
  const float* fcR_b1 = (const float*)d_in[7];
  const float* fcR_w2 = (const float*)d_in[8];
  const float* fcR_b2 = (const float*)d_in[9];
  const float* fcT1_w1 = (const float*)d_in[10];
  const float* fcT1_b1 = (const float*)d_in[11];
  const float* fcT1_w2 = (const float*)d_in[12];
  const float* fcT1_b2 = (const float*)d_in[13];
  const float* fcT2_w1 = (const float*)d_in[14];
  const float* fcT2_b1 = (const float*)d_in[15];
  const float* fcT2_w2 = (const float*)d_in[16];
  const float* fcT2_b2 = (const float*)d_in[17];

  char* ws = (char*)d_ws;
  size_t off = 0;
  _Float16* vi = (_Float16*)(ws + off); off += (size_t)NROW * NHID * sizeof(_Float16);
  _Float16* vk = (_Float16*)(ws + off); off += (size_t)NROW * NHID * sizeof(_Float16);
  _Float16* vm = (_Float16*)(ws + off); off += (size_t)NROW * NHID * sizeof(_Float16);
  float* Qg   = (float*)(ws + off); off += (size_t)NROW * NHID * sizeof(float);
  float* PsA  = (float*)(ws + off); off += (size_t)NROW * sizeof(float);
  float* PlA  = (float*)(ws + off); off += (size_t)NROW * sizeof(float);
  float* pk1  = (float*)(ws + off); off += (size_t)NSS * NHID * sizeof(float);
  float* pk2  = (float*)(ws + off); off += (size_t)NSS * NHID * sizeof(float);
  float* pk23 = (float*)(ws + off); off += (size_t)NSS * NHID * sizeof(float);
  float* pk3  = (float*)(ws + off); off += (size_t)NSS * NHID * sizeof(float);
  float* pgl  = (float*)(ws + off); off += (size_t)NSS * NHID * sizeof(float);
  float* pqb  = (float*)(ws + off); off += (size_t)NSS * NHID * sizeof(float);
  float* pge1 = (float*)(ws + off); off += (size_t)NSS * NHID * sizeof(float);
  float* pge2 = (float*)(ws + off); off += (size_t)NSS * NHID * sizeof(float);
  float* pga  = (float*)(ws + off); off += (size_t)NSS * NHID * sizeof(float);
  float* prw  = (float*)(ws + off); off += (size_t)NHID * NRT * NSS * sizeof(float);

  site_params_kernel<<<NSS, 256, 0, stream>>>(
      xc, fc_w1, fc_b1, fc_w2, fc_b2, fcR_w1, fcR_b1, fcR_w2, fcR_b2,
      pk1, pk2, pk23, pk3, pgl, pqb, pge1, pge2, pga, prw);

  pspl_kernel<<<(NROW + 255) / 256, 256, 0, stream>>>(x, PsA, PlA);

  dim3 g2((NROW / 16) / TPB_TILES, 2);
  mlp_big_kernel<<<g2, 256, 0, stream>>>(
      x, xc, fcT1_w1, fcT1_b1, fcT1_w2, fcT1_b2,
      fcT2_w1, fcT2_b1, fcT2_w2, fcT2_b2, vi, vk, vm);

  scan_kernel<<<(NSS * NHID + 255) / 256, 256, 0, stream>>>(
      x, PsA, PlA, vi, vk, vm,
      pk1, pk2, pk23, pk3, pgl, pqb, pge1, pge2, pga, Qg);

  conv_kernel<<<(NLL * NSS + 255) / 256, 256, 0, stream>>>(Qg, prw, (float*)d_out);
}
